// GNN_798863917760
// MI455X (gfx1250) — matrix operand order repaired and checked by reading, not compiled, not run on hardware
//
#include <hip/hip_runtime.h>
#include <cfloat>

#define NNODES 100000
#define SLOPE  0.2f

typedef __attribute__((ext_vector_type(16))) __bf16 v16bf;
typedef __attribute__((ext_vector_type(8)))  __bf16 v8bf;
typedef __attribute__((ext_vector_type(8)))  float  v8f;

__device__ __forceinline__ __bf16 f2bf(float f) {
    unsigned int u = __float_as_uint(f);
    u = u + 0x7fffu + ((u >> 16) & 1u);          // round-to-nearest-even
    unsigned short s = (unsigned short)(u >> 16);
    return __builtin_bit_cast(__bf16, s);
}
__device__ __forceinline__ unsigned int f2ord(float f) {
    unsigned int u = __float_as_uint(f);
    return (u & 0x80000000u) ? ~u : (u | 0x80000000u);
}
__device__ __forceinline__ float ord2f(unsigned int u) {
    u = (u & 0x80000000u) ? (u & 0x7fffffffu) : ~u;
    return __uint_as_float(u);
}
__device__ __forceinline__ float lrelu(float x) { return x > 0.f ? x : SLOPE * x; }

// ---------------------------------------------------------------------------
// W fp32 [128 x ndim] -> Wt bf16 [ndimPad x 128] (transposed, zero padded).
// Puts each WMMA B-fragment (K = kt+16*half+0..15 for column n) at 16
// contiguous bf16 -> one 32B load per lane per K step.
// ---------------------------------------------------------------------------
__global__ void gat_convert_w(const float* __restrict__ W,
                              __bf16* __restrict__ Wt, int ndim, int ndimPad)
{
    int t = blockIdx.x * blockDim.x + threadIdx.x;
    if (t >= ndimPad * 128) return;
    int n = t >> 7;            // /128
    int k = t & 127;
    float v = (n < ndim) ? W[(size_t)k * ndim + n] : 0.f;
    Wt[t] = f2bf(v);
}

// ---------------------------------------------------------------------------
// C[nrows x ndim] = A[nrows x 128] @ W[128 x ndim] via bf16 WMMA, f32 accum.
// Block = 256 threads = 8 waves. Block stages a 64x128 A tile in LDS (bf16,
// row stride 136 to avoid bank conflicts), one barrier, then each wave does
// 4 M-tiles x 4 K-steps = 16 unrolled v_wmma. EXEC stays full around WMMA
// (clamped indices / wave-uniform branches only).
// ---------------------------------------------------------------------------
#define AROWS 64
#define ASTR  136   // padded LDS row stride (bf16 elements)

__global__ void __launch_bounds__(256)
gat_gemm_wmma(const float* __restrict__ A, const __bf16* __restrict__ Wt,
              float* __restrict__ C, int nrows, int ndim)
{
    __shared__ __align__(16) __bf16 As[AROWS * ASTR];

    const int tid  = threadIdx.x;
    const int rowb = blockIdx.x * AROWS;

    // --- cooperative load + fp32->bf16 convert of 64x128 A tile ---
    {
        int row   = tid >> 2;            // 0..63
        int cbase = (tid & 3) * 32;      // 0,32,64,96
        int r = rowb + row; if (r >= nrows) r = nrows - 1;   // clamp (dup row)
        const float* __restrict__ ar = A + (size_t)r * 128 + cbase;
#pragma unroll
        for (int c8 = 0; c8 < 4; ++c8) {
            v8f  f  = *(const v8f*)(ar + c8 * 8);
            v8bf bv = __builtin_convertvector(f, v8bf);   // packed cvt
            *(v8bf*)(&As[row * ASTR + cbase + c8 * 8]) = bv;
        }
    }
    __syncthreads();

    const int lane = tid & 31;
    const int wave = tid >> 5;
    const int ndimPad = (ndim + 15) & ~15;
    if (wave * 16 >= ndimPad) return;    // wave-uniform (no barriers follow)

    const int m    = lane & 15;
    const int half = lane >> 4;
    const int n    = wave * 16 + m;      // < ndimPad (Wt zero-padded)

    const __bf16* __restrict__ wrow = Wt + (size_t)n * 128;

    v8f acc[4] = {{}, {}, {}, {}};
#pragma unroll
    for (int kt = 0; kt < 128; kt += 32) {
        // B fragment: element i <-> K = kt + 16*half + i, contiguous in Wt
        v16bf b = *(const v16bf*)(wrow + kt + 16 * half);
#pragma unroll
        for (int mt = 0; mt < 4; ++mt) {
            // A fragment: elem i <-> K = kt + 16*half + i
            const __bf16* ap = &As[(mt * 16 + m) * ASTR + kt + 16 * half];
            v8bf a0 = *(const v8bf*)ap;
            v8bf a1 = *(const v8bf*)(ap + 8);
            v16bf a = __builtin_shufflevector(a0, a1, 0, 1, 2, 3, 4, 5, 6, 7,
                                              8, 9, 10, 11, 12, 13, 14, 15);
            acc[mt] = __builtin_amdgcn_wmma_f32_16x16x32_bf16(
                          false, a, false, b, (short)0, acc[mt], false, false);
        }
    }
#pragma unroll
    for (int mt = 0; mt < 4; ++mt) {
#pragma unroll
        for (int rr = 0; rr < 8; ++rr) {
            int row = rowb + mt * 16 + rr + 8 * half;   // C: M = rr + 8*half
            if (row < nrows && n < ndim)
                C[(size_t)row * ndim + n] = acc[mt][rr];
        }
    }
}

// ---------------------------------------------------------------------------
// Per (node, head): attention dots, init segment-max with self-loop value,
// zero denom and aggregation buffer.
// ---------------------------------------------------------------------------
template <int H, int D>
__global__ void gat_alpha_init(const float* __restrict__ h,
                               const float* __restrict__ a_s,
                               const float* __restrict__ a_d,
                               float* __restrict__ alpha_s,
                               float* __restrict__ alpha_d,
                               unsigned int* __restrict__ mx,
                               float* __restrict__ dn,
                               float* __restrict__ agg,
                               int nnodes)
{
    int t = blockIdx.x * blockDim.x + threadIdx.x;
    if (t >= nnodes * H) return;
    int nidx = t / H, hd = t - nidx * H;
    const float* hp  = h + (size_t)nidx * (H * D) + hd * D;
    const float* asp = a_s + hd * D;
    const float* adp = a_d + hd * D;
    float* ag = agg + (size_t)nidx * (H * D) + hd * D;
    float ss = 0.f, sd = 0.f;
#pragma unroll
    for (int d = 0; d < D; ++d) {
        float v = hp[d];
        ss += v * asp[d];
        sd += v * adp[d];
        ag[d] = 0.f;
    }
    alpha_s[t] = ss;
    alpha_d[t] = sd;
    mx[t] = f2ord(lrelu(ss + sd));   // self-loop seeds the max
    dn[t] = 0.f;
}

// ---------------------------------------------------------------------------
// Pass 1 over real edges: segment max via order-preserving uint atomicMax.
// ---------------------------------------------------------------------------
template <int H>
__global__ void gat_edge_max(const long long* __restrict__ src,
                             const long long* __restrict__ dst,
                             const float* __restrict__ alpha_s,
                             const float* __restrict__ alpha_d,
                             unsigned int* __restrict__ mx,
                             long long nedges)
{
    long long t = (long long)blockIdx.x * blockDim.x + threadIdx.x;
    if (t >= nedges * H) return;
    long long e = t / H;                 // H constant -> shift
    int hd = (int)(t - e * H);
    int s = (int)src[e], d = (int)dst[e];
    float v = lrelu(alpha_s[s * H + hd] + alpha_d[d * H + hd]);
    atomicMax(&mx[d * H + hd], f2ord(v));
}

// ---------------------------------------------------------------------------
// Pass 2 over edges + self-loop tail: softmax denominator.
// ---------------------------------------------------------------------------
template <int H>
__global__ void gat_edge_sum(const long long* __restrict__ src,
                             const long long* __restrict__ dst,
                             const float* __restrict__ alpha_s,
                             const float* __restrict__ alpha_d,
                             const unsigned int* __restrict__ mx,
                             float* __restrict__ dn,
                             long long nedges, int nnodes)
{
    long long t = (long long)blockIdx.x * blockDim.x + threadIdx.x;
    long long tot = (nedges + nnodes) * (long long)H;
    if (t >= tot) return;
    long long e = t / H;
    int hd = (int)(t - e * H);
    int s, d;
    if (e < nedges) { s = (int)src[e]; d = (int)dst[e]; }
    else            { s = d = (int)(e - nedges); }
    float v = lrelu(alpha_s[s * H + hd] + alpha_d[d * H + hd]);
    float m = ord2f(mx[d * H + hd]);
    atomicAdd(&dn[d * H + hd], __expf(v - m));
}

// ---------------------------------------------------------------------------
// Pass 3 over edges + self-loop tail: scatter h[src] * alpha into agg[dst].
// One thread = 4 channels: float4 gather + 4 global_atomic_add_f32.
// ---------------------------------------------------------------------------
template <int H, int D>
__global__ void gat_edge_msg(const long long* __restrict__ src,
                             const long long* __restrict__ dst,
                             const float* __restrict__ alpha_s,
                             const float* __restrict__ alpha_d,
                             const unsigned int* __restrict__ mx,
                             const float* __restrict__ dn,
                             const float* __restrict__ h,
                             float* __restrict__ agg,
                             long long nedges, int nnodes)
{
    constexpr int CH4 = (H * D) / 4;
    long long t = (long long)blockIdx.x * blockDim.x + threadIdx.x;
    long long tot = (nedges + nnodes) * (long long)CH4;
    if (t >= tot) return;
    long long e = t / CH4;               // constant divisor
    int q = (int)(t - e * CH4);
    int c = q * 4;
    int hd = c / D;
    int s, d;
    if (e < nedges) { s = (int)src[e]; d = (int)dst[e]; }
    else            { s = d = (int)(e - nedges); }
    float v = lrelu(alpha_s[s * H + hd] + alpha_d[d * H + hd]);
    float m = ord2f(mx[d * H + hd]);
    float alpha = __expf(v - m) / dn[d * H + hd];
    const float4 hv = *(const float4*)(h + (size_t)s * (H * D) + c);
    float* ap = agg + (size_t)d * (H * D) + c;
    atomicAdd(ap + 0, hv.x * alpha);
    atomicAdd(ap + 1, hv.y * alpha);
    atomicAdd(ap + 2, hv.z * alpha);
    atomicAdd(ap + 3, hv.w * alpha);
}

// ---------------------------------------------------------------------------
// agg = (relu?)(agg + bias)   (in place; agg doubles as next-layer features)
// ---------------------------------------------------------------------------
template <int CH, int DO_RELU>
__global__ void gat_finalize(float* __restrict__ agg,
                             const float* __restrict__ bias, int nnodes)
{
    long long t = (long long)blockIdx.x * blockDim.x + threadIdx.x;
    if (t >= (long long)nnodes * CH) return;
    int c = (int)(t % CH);               // constant divisor
    float v = agg[t] + bias[c];
    if (DO_RELU) v = fmaxf(v, 0.f);
    agg[t] = v;
}

__global__ void gat_logsoftmax(const float* __restrict__ z,
                               float* __restrict__ out, int nnodes)
{
    constexpr int CH = 40;
    int n = blockIdx.x * blockDim.x + threadIdx.x;
    if (n >= nnodes) return;
    const float* zp = z + (size_t)n * CH;
    float m = -FLT_MAX;
#pragma unroll
    for (int c = 0; c < CH; ++c) m = fmaxf(m, zp[c]);
    float s = 0.f;
#pragma unroll
    for (int c = 0; c < CH; ++c) s += __expf(zp[c] - m);
    float ls = m + __logf(s);
    float* op = out + (size_t)n * CH;
#pragma unroll
    for (int c = 0; c < CH; ++c) op[c] = zp[c] - ls;
}

// ---------------------------------------------------------------------------
static inline unsigned blks(long long tot) { return (unsigned)((tot + 255) / 256); }

extern "C" void kernel_launch(void* const* d_in, const int* in_sizes, int n_in,
                              void* d_out, int out_size, void* d_ws, size_t ws_size,
                              hipStream_t stream)
{
    const int N = NNODES;
    const float* x = (const float*)d_in[0];
    const long long* ei = (const long long*)d_in[1];
    const long long E = (long long)in_sizes[1] / 2;
    const long long* src = ei;
    const long long* dst = ei + E;

    const float* W[4]  = {(const float*)d_in[2],  (const float*)d_in[6],
                          (const float*)d_in[10], (const float*)d_in[14]};
    const float* AS[4] = {(const float*)d_in[3],  (const float*)d_in[7],
                          (const float*)d_in[11], (const float*)d_in[15]};
    const float* AD[4] = {(const float*)d_in[4],  (const float*)d_in[8],
                          (const float*)d_in[12], (const float*)d_in[16]};
    const float* BI[4] = {(const float*)d_in[5],  (const float*)d_in[9],
                          (const float*)d_in[13], (const float*)d_in[17]};

    // Workspace: 2 ping-pong feature buffers + h + small arrays + bf16 W.
    float* buf0 = (float*)d_ws;
    float* buf1 = buf0 + (size_t)N * 128;
    float* hbuf = buf1 + (size_t)N * 128;
    float* as_  = hbuf + (size_t)N * 128;
    float* ad_  = as_  + (size_t)N * 4;
    unsigned int* mx = (unsigned int*)(ad_ + (size_t)N * 4);
    float* dn = (float*)(mx + (size_t)N * 4);
    __bf16* wt = (__bf16*)(dn + (size_t)N * 4);

    const float* in_feat = x;
    float* aggs[4] = {buf0, buf1, buf0, buf1};

    for (int L = 0; L < 4; ++L) {
        const int H = (L == 3) ? 1 : 4;
        const int D = (L == 3) ? 40 : 32;
        const int CH = H * D;
        const int ndimPad = (CH + 15) & ~15;
        float* agg = aggs[L];

        gat_convert_w<<<blks(ndimPad * 128), 256, 0, stream>>>(W[L], wt, CH, ndimPad);
        gat_gemm_wmma<<<(N + AROWS - 1) / AROWS, 256, 0, stream>>>(
            in_feat, wt, hbuf, N, CH);

        if (L < 3) {
            gat_alpha_init<4, 32><<<blks((long long)N * 4), 256, 0, stream>>>(
                hbuf, AS[L], AD[L], as_, ad_, mx, dn, agg, N);
            gat_edge_max<4><<<blks(E * 4), 256, 0, stream>>>(src, dst, as_, ad_, mx, E);
            gat_edge_sum<4><<<blks((E + N) * 4), 256, 0, stream>>>(
                src, dst, as_, ad_, mx, dn, E, N);
            gat_edge_msg<4, 32><<<blks((E + N) * 32), 256, 0, stream>>>(
                src, dst, as_, ad_, mx, dn, hbuf, agg, E, N);
            gat_finalize<128, 1><<<blks((long long)N * 128), 256, 0, stream>>>(
                agg, BI[L], N);
        } else {
            gat_alpha_init<1, 40><<<blks((long long)N), 256, 0, stream>>>(
                hbuf, AS[L], AD[L], as_, ad_, mx, dn, agg, N);
            gat_edge_max<1><<<blks(E), 256, 0, stream>>>(src, dst, as_, ad_, mx, E);
            gat_edge_sum<1><<<blks(E + N), 256, 0, stream>>>(
                src, dst, as_, ad_, mx, dn, E, N);
            gat_edge_msg<1, 40><<<blks((E + N) * 10), 256, 0, stream>>>(
                src, dst, as_, ad_, mx, dn, hbuf, agg, E, N);
            gat_finalize<40, 0><<<blks((long long)N * 40), 256, 0, stream>>>(
                agg, BI[L], N);
        }
        in_feat = agg;
    }

    gat_logsoftmax<<<blks(N), 256, 0, stream>>>(aggs[3], (float*)d_out, N);
}
